// AttentionWithTime_16398185136216
// MI455X (gfx1250) — compile-verified, matrix operand order checked by reading, not run on hardware
//
#include <hip/hip_runtime.h>
#include <cstdint>

// ---------------------------------------------------------------------------
// AttentionWithTime for MI455X (gfx1250, wave32, WMMA).
// ~73 GFLOP, ~120MB working set (fits in 192MB L2) -> matrix-core bound.
// All GEMM-shaped math goes through v_wmma_f32_16x16x32_f16 (f16 in, f32 acc).
// Contiguous global->LDS staging uses the CDNA5 async path
// (global_load_async_to_lds_b128 + s_wait_asynccnt), which bypasses VGPRs and
// overlaps with WMMA issue. Attention is fused flash-style (exp kernel, no
// softmax -> no running max), never materializing the NxN matrix.
// t_shift cancels exactly inside feature_norm over the sequence dim, so only
// t_scale is applied (sigmoid gate in the Wf1 GEMM epilogue).
// ---------------------------------------------------------------------------

#define B_SZ     8
#define N_SEQ    1024
#define D_MODEL  256
#define H_HEADS  8
#define D_HEAD   64
#define D_QKV    3072      // 2*512 + 2048
#define D_EXPAND 1024
#define D_TIME   256
#define TS_COLS  (D_MODEL + D_EXPAND)   // 1280
#define ROWS     (B_SZ * N_SEQ)         // 8192
#define EPSN     1e-5f

typedef __attribute__((ext_vector_type(16))) _Float16 v16h;
typedef __attribute__((ext_vector_type(8)))  float    v8f;

__device__ __forceinline__ float sigmoidf_(float x) { return 1.f / (1.f + __expf(-x)); }
__device__ __forceinline__ float swishf_(float x)   { return x * sigmoidf_(x); }

__device__ __forceinline__ v8f wmma_f16f32(v16h a, v16h b, v8f c) {
  // (neg_a, A, neg_b, B, c_mod, C, reuse_a, reuse_b)
  return __builtin_amdgcn_wmma_f32_16x16x32_f16(false, a, false, b, (short)0, c,
                                                false, false);
}

// CDNA5 async global->LDS 16B copy (per-lane), tracked by ASYNCcnt.
// lds_addr: low 32 bits of a generic shared pointer == wave LDS byte offset.
__device__ __forceinline__ void async_copy_b128(void* lds_dst, const void* gsrc) {
  unsigned lds_addr = (unsigned)(uintptr_t)lds_dst;
  uint64_t ga = (uint64_t)(uintptr_t)gsrc;
  asm volatile("global_load_async_to_lds_b128 %0, %1, off"
               :: "v"(lds_addr), "v"(ga) : "memory");
}
__device__ __forceinline__ void wait_async0() {
  asm volatile("s_wait_asynccnt 0" ::: "memory");
}

// Load a 16x32 f16 fragment in the CDNA5 A-matrix layout (also used for B via
// the symmetric column layout). p -> element (0,0), ld = row stride in halves.
// Lane L (m=L&15, hi=L>>4) holds row m; half i holds k=(i&7)+hi*8+(i>>3)*16.
__device__ __forceinline__ v16h load_frag16(const _Float16* p, int ld) {
  int lane = threadIdx.x & 31;
  const _Float16* r = p + (lane & 15) * ld + (lane >> 4) * 8;
  v16h f;
#pragma unroll
  for (int i = 0; i < 8; ++i) f[i] = r[i];
#pragma unroll
  for (int i = 0; i < 8; ++i) f[8 + i] = r[16 + i];
  return f;
}

// ---------------------------------------------------------------------------
// fp32 -> f16 weight conversion
__global__ void cvt16_kernel(const float* __restrict__ src,
                             _Float16* __restrict__ dst, int n) {
  int i = blockIdx.x * blockDim.x + threadIdx.x;
  if (i < n) dst[i] = (_Float16)src[i];
}

// Per-(b,d) mean / rstd over the sequence dim (population variance).
__global__ __launch_bounds__(D_MODEL) void colstats_kernel(
    const float* __restrict__ x, float* __restrict__ mean,
    float* __restrict__ rstd) {
  int b = blockIdx.x, d = threadIdx.x;
  const float* p = x + (size_t)b * N_SEQ * D_MODEL + d;
  float s = 0.f, s2 = 0.f;
  for (int n = 0; n < N_SEQ; ++n) {
    float v = p[(size_t)n * D_MODEL];
    s += v; s2 += v * v;
  }
  float m   = s * (1.f / N_SEQ);
  float var = s2 * (1.f / N_SEQ) - m * m;
  mean[b * D_MODEL + d] = m;
  rstd[b * D_MODEL + d] = rsqrtf(var + EPSN);
}

// out = f16( swish( gamma*(x-mean)*rstd + beta [+ pos] ) )
__global__ __launch_bounds__(D_MODEL) void pack_norm_swish_kernel(
    const float* __restrict__ x, const float* __restrict__ mean,
    const float* __restrict__ rstd, const float* __restrict__ gamma,
    const float* __restrict__ beta, const float* __restrict__ pos,
    _Float16* __restrict__ out) {
  int row = blockIdx.x;                 // b*N + n
  int b = row >> 10, n = row & (N_SEQ - 1);
  int d = threadIdx.x;
  size_t idx = (size_t)row * D_MODEL + d;
  float v = gamma[d] * (x[idx] - mean[b * D_MODEL + d]) * rstd[b * D_MODEL + d]
          + beta[d];
  if (pos) v += pos[n * D_MODEL + d];
  out[idx] = (_Float16)swishf_(v);
}

// ts = t @ Wt + bt   (8 x 1280, K=256 -- tiny, scalar kernel)
__global__ void ts_kernel(const float* __restrict__ t,
                          const float* __restrict__ Wt,
                          const float* __restrict__ bt,
                          float* __restrict__ ts) {
  int i = blockIdx.x * blockDim.x + threadIdx.x;
  if (i >= B_SZ * TS_COLS) return;
  int b = i / TS_COLS, j = i % TS_COLS;
  float acc = bt[j];
  const float* tb = t + b * D_TIME;
  for (int k = 0; k < D_TIME; ++k) acc += tb[k] * Wt[k * TS_COLS + j];
  ts[i] = acc;
}

// q2[b,h,n] = |q|^2, k2[b,h,n] = |k|^2 from packed f16 qkv.
__global__ void q2k2_kernel(const _Float16* __restrict__ qkv,
                            float* __restrict__ q2, float* __restrict__ k2) {
  int i = blockIdx.x * blockDim.x + threadIdx.x;   // over B*N*H = 65536
  if (i >= B_SZ * N_SEQ * H_HEADS) return;
  int b = i >> 13; int rem = i & 8191; int n = rem >> 3; int h = rem & 7;
  const _Float16* base = qkv + (size_t)(b * N_SEQ + n) * D_QKV + h * D_HEAD;
  float sq = 0.f, sk = 0.f;
  for (int d = 0; d < D_HEAD; ++d) {
    float qv = (float)base[d];
    float kv = (float)base[512 + d];
    sq += qv * qv; sk += kv * kv;
  }
  int bh = b * H_HEADS + h;
  q2[(size_t)bh * N_SEQ + n] = sq;
  k2[(size_t)bh * N_SEQ + n] = sk;
}

// ---------------------------------------------------------------------------
// Generic f16 WMMA GEMM: out = epilogue(A[M,K] @ B[K,N] + bias)
// Block: 256 thr = 8 waves (4x2), wave tile 32x64, block tile 128x128,
// K-step 32 -> 8 WMMAs per K-step per wave vs 6 fragment loads.
// A tile staged via async global->LDS (contiguous); B tile staged manually
// (transposed to [n][k] for contiguous fragment reads).
// Epilogue: optional bias, per-(b,col) sigmoid gate, swish, f32 residual,
// f32 and/or f16 outputs. M%128==0, N%128==0, K%32==0 (true for all calls).
__global__ __launch_bounds__(256) void gemm16_kernel(
    const _Float16* __restrict__ A, const _Float16* __restrict__ Bm,
    int M, int Nn, int K,
    const float* __restrict__ bias, const float* __restrict__ resid,
    const float* __restrict__ gate, int gate_ld, int rows_per_b, int act,
    float* __restrict__ outF, _Float16* __restrict__ outH) {
  __shared__ __align__(16) _Float16 sA[128 * 32];   // [m][k]
  __shared__ __align__(16) _Float16 sB[128 * 32];   // [n][k] (transposed)
  int nBase   = blockIdx.x * 128;
  int rowBase = blockIdx.y * 128;
  int wave = threadIdx.x >> 5, lane = threadIdx.x & 31;
  int wr = wave >> 1, wc = wave & 1;
  int m16 = lane & 15, hi = lane >> 4;

  v8f zero = {0.f, 0.f, 0.f, 0.f, 0.f, 0.f, 0.f, 0.f};
  v8f acc[2][4];
#pragma unroll
  for (int i = 0; i < 2; ++i)
#pragma unroll
    for (int j = 0; j < 4; ++j) acc[i][j] = zero;

  int nk = K >> 5;
  for (int kt = 0; kt < nk; ++kt) {
    __syncthreads();
    {
      // A tile: 128 rows x 32 halves = 512 x 16B chunks; 2 async copies/thread.
      // chunk c -> row c>>2, 16B-part c&3; LDS offset c*16 (row-major [128][32]).
#pragma unroll
      for (int u = 0; u < 2; ++u) {
        int c = threadIdx.x * 2 + u;
        const _Float16* src =
            A + (size_t)(rowBase + (c >> 2)) * K + kt * 32 + (c & 3) * 8;
        async_copy_b128(&sA[c * 8], src);
      }
    }
    {
      // B tile: 32 rows(k) x 128 halves(n), transposed scatter into sB[n][k].
      int kk = threadIdx.x >> 3, g = threadIdx.x & 7;    // g: 16-col group
      const _Float16* src = Bm + (size_t)(kt * 32 + kk) * Nn + nBase + g * 16;
      _Float16 tmp[16];
      *(uint4*)&tmp[0] = ((const uint4*)src)[0];
      *(uint4*)&tmp[8] = ((const uint4*)src)[1];
#pragma unroll
      for (int j = 0; j < 16; ++j) sB[(g * 16 + j) * 32 + kk] = tmp[j];
      if (kt + 1 < nk) __builtin_prefetch(src + (size_t)32 * Nn, 0, 1);
    }
    wait_async0();
    __syncthreads();

    v16h af0 = load_frag16(&sA[(wr * 32 + 0)  * 32], 32);
    v16h af1 = load_frag16(&sA[(wr * 32 + 16) * 32], 32);
#pragma unroll
    for (int tn = 0; tn < 4; ++tn) {
      v16h bf = load_frag16(&sB[(wc * 64 + tn * 16) * 32], 32);
      acc[0][tn] = wmma_f16f32(af0, bf, acc[0][tn]);
      acc[1][tn] = wmma_f16f32(af1, bf, acc[1][tn]);
    }
  }

  // Epilogue. C layout: VGPR r -> rows {r, r+8} (lane<16 / lane>=16), col=lane&15
#pragma unroll
  for (int tm = 0; tm < 2; ++tm)
#pragma unroll
    for (int tn = 0; tn < 4; ++tn) {
      int col = nBase + wc * 64 + tn * 16 + m16;
#pragma unroll
      for (int r = 0; r < 8; ++r) {
        int row = rowBase + wr * 32 + tm * 16 + r + hi * 8;
        float v = acc[tm][tn][r];
        if (bias)  v += bias[col];
        if (gate)  v *= sigmoidf_(gate[(row / rows_per_b) * gate_ld + col]);
        if (act)   v = swishf_(v);
        if (resid) v += resid[(size_t)row * Nn + col];
        size_t o = (size_t)row * Nn + col;
        if (outF) outF[o] = v;
        if (outH) outH[o] = (_Float16)v;
      }
    }
}

// ---------------------------------------------------------------------------
// Fused Gaussian-kernel attention (no softmax => no running max needed).
// Grid: (N/128, B*H). Block: 256 thr = 8 waves; each wave owns 16 query rows
// and accumulates a full 16x256 output in registers (16 C-tiles = 128 VGPRs).
// Per 32-key block: K tile async-staged to LDS, V tile transposed manually,
// S = Q K^T via WMMA, exp epilogue in regs, S->P relayout via 1KB per-wave
// LDS bounce (C-tile layout -> A-fragment layout), P@V via WMMA.
__global__ __launch_bounds__(256) void attn_kernel(
    const _Float16* __restrict__ qkv, const float* __restrict__ q2,
    const float* __restrict__ k2, const float* __restrict__ scale,
    _Float16* __restrict__ outH) {
  __shared__ __align__(16) _Float16 sK[32 * 64];    // [key][dh]
  __shared__ __align__(16) _Float16 sV[256 * 32];   // [col][key] (transposed)
  __shared__ __align__(16) _Float16 sS[8][16 * 32]; // per-wave P scratch

  int bh = blockIdx.y; int b = bh >> 3, h = bh & 7;
  int wave = threadIdx.x >> 5, lane = threadIdx.x & 31;
  int m16 = lane & 15, hi = lane >> 4;
  int qbase = blockIdx.x * 128 + wave * 16;

  float sc = scale[0];
  float inv_s2 = 1.f / (sc * sc);
  const _Float16* base = qkv + (size_t)b * N_SEQ * D_QKV;

  // Q fragments (16 queries x 64 dh = two K-steps), loaded once.
  v16h qf[2];
  {
    const _Float16* qrow =
        base + (size_t)(qbase + m16) * D_QKV + h * D_HEAD + hi * 8;
#pragma unroll
    for (int ks = 0; ks < 2; ++ks) {
#pragma unroll
      for (int i = 0; i < 8; ++i) qf[ks][i]     = qrow[ks * 32 + i];
#pragma unroll
      for (int i = 0; i < 8; ++i) qf[ks][8 + i] = qrow[ks * 32 + 16 + i];
    }
  }
  float q2v[8];
#pragma unroll
  for (int r = 0; r < 8; ++r)
    q2v[r] = q2[(size_t)bh * N_SEQ + qbase + r + hi * 8];

  v8f zero = {0.f, 0.f, 0.f, 0.f, 0.f, 0.f, 0.f, 0.f};
  v8f acc[16];
#pragma unroll
  for (int i = 0; i < 16; ++i) acc[i] = zero;

  for (int kb = 0; kb < N_SEQ / 32; ++kb) {
    __syncthreads();
    {
      // K tile (32 keys x 64 dh halves): 256 x 16B chunks, async to LDS.
      int kk = threadIdx.x >> 3, g = threadIdx.x & 7;
      const _Float16* ksrc =
          base + (size_t)(kb * 32 + kk) * D_QKV + 512 + h * D_HEAD + g * 8;
      async_copy_b128(&sK[threadIdx.x * 8], ksrc);
      // V tile (32 keys x 256 cols), transposed into sV[col][key].
      const _Float16* vsrc =
          base + (size_t)(kb * 32 + kk) * D_QKV + 1024 + h * D_MODEL + g * 32;
#pragma unroll
      for (int j = 0; j < 32; ++j) sV[(g * 32 + j) * 32 + kk] = vsrc[j];
    }
    wait_async0();
    __syncthreads();

    // S tiles for key halves [kb*32, +16) and [kb*32+16, +32).
    v8f s0 = zero, s1 = zero;
#pragma unroll
    for (int ks = 0; ks < 2; ++ks) {
      v16h bk0 = load_frag16(sK + ks * 32, 64);
      s0 = wmma_f16f32(qf[ks], bk0, s0);
      v16h bk1 = load_frag16(sK + 16 * 64 + ks * 32, 64);
      s1 = wmma_f16f32(qf[ks], bk1, s1);
    }

    float k20 = k2[(size_t)bh * N_SEQ + kb * 32 + m16];
    float k21 = k2[(size_t)bh * N_SEQ + kb * 32 + 16 + m16];
    _Float16* ss = sS[wave];
#pragma unroll
    for (int r = 0; r < 8; ++r) {
      int row = r + hi * 8;
      float d0 = q2v[r] + k20 - 2.f * s0[r];
      float d1 = q2v[r] + k21 - 2.f * s1[r];
      ss[row * 32 + m16]      = (_Float16)__expf(-fmaxf(d0, 0.f) * inv_s2);
      ss[row * 32 + 16 + m16] = (_Float16)__expf(-fmaxf(d1, 0.f) * inv_s2);
    }
    // Per-wave LDS RAW: only this wave touches ss; DS ops are wave-ordered,
    // wait for store completion before the fragment reload.
    asm volatile("s_wait_dscnt 0" ::: "memory");
    v16h pf = load_frag16(ss, 32);

#pragma unroll
    for (int ct = 0; ct < 16; ++ct) {
      v16h bv = load_frag16(sV + ct * 16 * 32, 32);
      acc[ct] = wmma_f16f32(pf, bv, acc[ct]);
    }
  }

  // out[b, n, h*256 + c] (matches reference transpose+reshape), f16 for Wm GEMM.
#pragma unroll
  for (int ct = 0; ct < 16; ++ct) {
#pragma unroll
    for (int r = 0; r < 8; ++r) {
      int row = qbase + r + hi * 8;
      int col = ct * 16 + m16;
      outH[(size_t)(b * N_SEQ + row) * (H_HEADS * D_MODEL) + h * D_MODEL + col] =
          (_Float16)acc[ct][r];
    }
  }
}

// ---------------------------------------------------------------------------
extern "C" void kernel_launch(void* const* d_in, const int* in_sizes, int n_in,
                              void* d_out, int out_size, void* d_ws,
                              size_t ws_size, hipStream_t stream) {
  (void)in_sizes; (void)n_in; (void)out_size; (void)ws_size;
  const float* x      = (const float*)d_in[0];
  const float* t      = (const float*)d_in[1];
  const float* gamma  = (const float*)d_in[2];
  const float* beta   = (const float*)d_in[3];
  const float* pos    = (const float*)d_in[4];
  const float* Wqkv   = (const float*)d_in[5];
  const float* bqkv   = (const float*)d_in[6];
  const float* Wm     = (const float*)d_in[7];
  const float* bm     = (const float*)d_in[8];
  const float* Wt     = (const float*)d_in[9];
  const float* bt     = (const float*)d_in[10];
  const float* gamma1 = (const float*)d_in[11];
  const float* beta1  = (const float*)d_in[12];
  const float* Wf1    = (const float*)d_in[13];
  const float* bf1    = (const float*)d_in[14];
  const float* Wf2    = (const float*)d_in[15];
  const float* bf2    = (const float*)d_in[16];
  const float* scale  = (const float*)d_in[17];
  float* out = (float*)d_out;

  char* ws = (char*)d_ws;
  size_t off = 0;
  auto alloc = [&](size_t bytes) -> void* {
    void* p = ws + off;
    off = (off + bytes + 255) & ~(size_t)255;
    return p;
  };
  float*    mean0  = (float*)alloc(B_SZ * D_MODEL * 4);
  float*    rstd0  = (float*)alloc(B_SZ * D_MODEL * 4);
  float*    mean1  = (float*)alloc(B_SZ * D_MODEL * 4);
  float*    rstd1  = (float*)alloc(B_SZ * D_MODEL * 4);
  _Float16* h16    = (_Float16*)alloc((size_t)ROWS * D_MODEL * 2);
  _Float16* Wqkv16 = (_Float16*)alloc((size_t)D_MODEL * D_QKV * 2);
  _Float16* Wm16   = (_Float16*)alloc((size_t)2048 * 256 * 2);
  _Float16* Wf116  = (_Float16*)alloc((size_t)256 * 1024 * 2);
  _Float16* Wf216  = (_Float16*)alloc((size_t)1024 * 256 * 2);
  _Float16* qkv16  = (_Float16*)alloc((size_t)ROWS * D_QKV * 2);
  float*    q2     = (float*)alloc((size_t)B_SZ * H_HEADS * N_SEQ * 4);
  float*    k2     = (float*)alloc((size_t)B_SZ * H_HEADS * N_SEQ * 4);
  _Float16* attn16 = (_Float16*)alloc((size_t)ROWS * 2048 * 2);
  float*    x1     = (float*)alloc((size_t)ROWS * D_MODEL * 4);
  _Float16* h2_16  = (_Float16*)alloc((size_t)ROWS * D_MODEL * 2);
  _Float16* ff16   = (_Float16*)alloc((size_t)ROWS * D_EXPAND * 2);
  float*    tsbuf  = (float*)alloc((size_t)B_SZ * TS_COLS * 4);

  // Weight conversions (f32 -> f16)
  cvt16_kernel<<<(D_MODEL * D_QKV) / 256, 256, 0, stream>>>(Wqkv, Wqkv16,
                                                            D_MODEL * D_QKV);
  cvt16_kernel<<<(2048 * 256) / 256, 256, 0, stream>>>(Wm, Wm16, 2048 * 256);
  cvt16_kernel<<<(256 * 1024) / 256, 256, 0, stream>>>(Wf1, Wf116, 256 * 1024);
  cvt16_kernel<<<(1024 * 256) / 256, 256, 0, stream>>>(Wf2, Wf216, 1024 * 256);

  // h = swish(featnorm(x) + pos), packed f16
  colstats_kernel<<<B_SZ, D_MODEL, 0, stream>>>(x, mean0, rstd0);
  pack_norm_swish_kernel<<<ROWS, D_MODEL, 0, stream>>>(x, mean0, rstd0, gamma,
                                                       beta, pos, h16);

  // qkv = h @ Wqkv + bqkv (f16 out)
  gemm16_kernel<<<dim3(D_QKV / 128, ROWS / 128), 256, 0, stream>>>(
      h16, Wqkv16, ROWS, D_QKV, D_MODEL, bqkv, nullptr, nullptr, 1, 1, 0,
      nullptr, qkv16);

  // row norms + time projection (independent, small)
  q2k2_kernel<<<(B_SZ * N_SEQ * H_HEADS) / 256, 256, 0, stream>>>(qkv16, q2, k2);
  ts_kernel<<<(B_SZ * TS_COLS + 255) / 256, 256, 0, stream>>>(t, Wt, bt, tsbuf);

  // fused exp(-d2/s^2) attention
  attn_kernel<<<dim3(N_SEQ / 128, B_SZ * H_HEADS), 256, 0, stream>>>(
      qkv16, q2, k2, scale, attn16);

  // x1 = x + attn_out @ Wm + bm (f32 out)
  gemm16_kernel<<<dim3(256 / 128, ROWS / 128), 256, 0, stream>>>(
      attn16, Wm16, ROWS, 256, 2048, bm, x, nullptr, 1, 1, 0, x1, nullptr);

  // h2 = swish(featnorm1(x1))  [t_shift cancels inside featnorm]
  colstats_kernel<<<B_SZ, D_MODEL, 0, stream>>>(x1, mean1, rstd1);
  pack_norm_swish_kernel<<<ROWS, D_MODEL, 0, stream>>>(x1, mean1, rstd1, gamma1,
                                                       beta1, nullptr, h2_16);

  // ff = swish((h2 @ Wf1 + bf1) * sigmoid(t_scale)) (f16 out)
  gemm16_kernel<<<dim3(D_EXPAND / 128, ROWS / 128), 256, 0, stream>>>(
      h2_16, Wf116, ROWS, D_EXPAND, D_MODEL, bf1, nullptr, tsbuf + D_MODEL,
      TS_COLS, N_SEQ, 1, nullptr, ff16);

  // out = ff @ Wf2 + bf2 + x1 (f32 -> d_out)
  gemm16_kernel<<<dim3(256 / 128, ROWS / 128), 256, 0, stream>>>(
      ff16, Wf216, ROWS, 256, D_EXPAND, bf2, x1, nullptr, 1, 1, 0, out,
      nullptr);
}
